// SetAttention_2628519985689
// MI455X (gfx1250) — compile-verified, matrix operand order checked by reading, not run on hardware
//
#include <hip/hip_runtime.h>
#include <math.h>

typedef __bf16 bf16;
typedef bf16 v16bf __attribute__((ext_vector_type(16)));
typedef bf16 v8bf  __attribute__((ext_vector_type(8)));
typedef float v8f  __attribute__((ext_vector_type(8)));

// ---------------- problem constants ----------------
#define BB   4
#define HH   64
#define WW   256
#define CC   256
#define NHD  8
#define HSP  4
#define WSP  8
#define LL   (HH * WW)            // 16384
#define MTOK (BB * LL)            // 65536 token rows
#define HD   (CC / NHD)           // 32
#define NTOK 32                   // tokens per window
#define NWIN (BB * (HH / HSP) * (WW / WSP))  // 2048
#define FF   (4 * CC)             // 1024
#define EPSV 1e-5f
#define SCL  0.17677669529663687f // 32^-0.5

// ---------------- helpers ----------------
__device__ __forceinline__ bf16 to_bf16(float f) {
  unsigned u = __builtin_bit_cast(unsigned, f);
  unsigned r = u + 0x7FFFu + ((u >> 16) & 1u);   // round-to-nearest-even
  unsigned short h = (unsigned short)(r >> 16);
  return __builtin_bit_cast(bf16, h);
}

__device__ __forceinline__ v8f wmma_bf16(v16bf a, v16bf b, v8f c) {
  // D = A(16x32 bf16) x B(32x16 bf16) + C(16x16 f32)
  return __builtin_amdgcn_wmma_f32_16x16x32_bf16(false, a, false, b, (short)0, c,
                                                 false, false);
}

// load one 16-element fragment: 8 bf16 at p, 8 bf16 at p+16 (A/B K-split layout)
__device__ __forceinline__ v16bf frag_ld(const bf16* p) {
  v8bf lo = *(const v8bf*)p;
  v8bf hi = *(const v8bf*)(p + 16);
  return __builtin_shufflevector(lo, hi, 0, 1, 2, 3, 4, 5, 6, 7,
                                 8, 9, 10, 11, 12, 13, 14, 15);
}

// window w, token n  ->  global token row in (B, L) flattening
__device__ __forceinline__ int win_row(int w, int n) {
  int b   = w >> 9;        // / 512 windows per batch
  int rem = w & 511;
  int hb  = rem >> 5;      // / 32
  int wb  = rem & 31;
  int hs  = n >> 3;
  int ws  = n & 7;
  int l   = (hb * HSP + hs) * WW + wb * WSP + ws;
  return (b << 14) + l;    // b * 16384 + l
}

// ---------------- fp32 -> bf16 weight conversion ----------------
__global__ void cvt_kernel(const float* __restrict__ s, bf16* __restrict__ d, int n) {
  int i = blockIdx.x * 256 + threadIdx.x;
  if (i < n) d[i] = to_bf16(s[i]);
}

// ---------------- LayerNorm (one wave per 256-wide token row) ----------------
__global__ void ln_kernel(const float* __restrict__ x, const float* __restrict__ g,
                          const float* __restrict__ b, bf16* __restrict__ y) {
  int token = blockIdx.x * 8 + (threadIdx.x >> 5);
  int lane  = threadIdx.x & 31;
  const float* row = x + (size_t)token * CC + lane * 8;
  float4 f0 = *(const float4*)row;
  float4 f1 = *(const float4*)(row + 4);
  float vals[8] = {f0.x, f0.y, f0.z, f0.w, f1.x, f1.y, f1.z, f1.w};
  float s = 0.f, sq = 0.f;
  for (int i = 0; i < 8; ++i) { s += vals[i]; sq += vals[i] * vals[i]; }
  for (int m = 1; m < 32; m <<= 1) {
    s  += __shfl_xor(s,  m, 32);
    sq += __shfl_xor(sq, m, 32);
  }
  float mean = s * (1.0f / CC);
  float var  = sq * (1.0f / CC) - mean * mean;
  float rs   = rsqrtf(var + EPSV);
  bf16* out = y + (size_t)token * CC + lane * 8;
  int col = lane * 8;
  for (int i = 0; i < 8; ++i)
    out[i] = to_bf16((vals[i] - mean) * rs * g[col + i] + b[col + i]);
}

// ---------------- generic bf16 WMMA GEMM:  out = epi(A * Bw^T + bias) ----------------
// A: MxK row-major bf16.  Bw: NxK row-major bf16 (weight layout -> WMMA B loaded
// column-major, which is exactly K-contiguous rows of Bw).
// Block tile 128x128: 8 waves in a 2x4 grid, each computing 64x32 via 4x2 WMMA
// tiles (8 WMMAs per wave per 32-wide K-step).
template <bool GELU, bool RES, bool OUTBF>
__global__ void gemm_bf16_kernel(const bf16* __restrict__ A, const bf16* __restrict__ Bw,
                                 const float* __restrict__ bias,
                                 const float* __restrict__ res,
                                 float* __restrict__ outf, bf16* __restrict__ outb,
                                 int M, int N, int K, float scale) {
  __shared__ __align__(16) bf16 As[128 * 40];
  __shared__ __align__(16) bf16 Bs[128 * 40];
  const int t    = threadIdx.x;
  const int wave = t >> 5, lane = t & 31;
  const int wr = wave >> 2, wc = wave & 3;        // 2x4 wave grid
  const int mBase = blockIdx.y * 128;
  const int nBase = blockIdx.x * 128;
  const int ln = lane & 15, hi = lane >> 4, kb = hi << 3;

  v8f acc[4][2];
  for (int i = 0; i < 4; ++i)
    for (int j = 0; j < 2; ++j)
      for (int v = 0; v < 8; ++v) acc[i][j][v] = 0.0f;

  // cooperative loaders: each thread moves 16 contiguous bf16 of A and of B
  const int lr = t >> 1, lc = (t & 1) * 16;
  for (int kk = 0; kk < K; kk += 32) {
    *(v8bf*)&As[lr * 40 + lc]     = *(const v8bf*)&A [(size_t)(mBase + lr) * K + kk + lc];
    *(v8bf*)&As[lr * 40 + lc + 8] = *(const v8bf*)&A [(size_t)(mBase + lr) * K + kk + lc + 8];
    *(v8bf*)&Bs[lr * 40 + lc]     = *(const v8bf*)&Bw[(size_t)(nBase + lr) * K + kk + lc];
    *(v8bf*)&Bs[lr * 40 + lc + 8] = *(const v8bf*)&Bw[(size_t)(nBase + lr) * K + kk + lc + 8];
    if (kk + 32 < K) {   // pull next K-slice toward the caches early
      __builtin_prefetch(&A [(size_t)(mBase + lr) * K + kk + 32 + lc], 0, 0);
      __builtin_prefetch(&Bw[(size_t)(nBase + lr) * K + kk + 32 + lc], 0, 0);
    }
    __syncthreads();

    v16bf af[4], bfr[2];
    for (int mi = 0; mi < 4; ++mi)
      af[mi] = frag_ld(&As[(wr * 64 + mi * 16 + ln) * 40 + kb]);
    for (int nj = 0; nj < 2; ++nj)
      bfr[nj] = frag_ld(&Bs[(wc * 32 + nj * 16 + ln) * 40 + kb]);

    for (int mi = 0; mi < 4; ++mi)
      for (int nj = 0; nj < 2; ++nj)
        acc[mi][nj] = wmma_bf16(af[mi], bfr[nj], acc[mi][nj]);
    __syncthreads();
  }

  // epilogue: + bias, (gelu), *scale, (+residual), store bf16 or f32
  for (int mi = 0; mi < 4; ++mi)
    for (int nj = 0; nj < 2; ++nj) {
      int n = nBase + wc * 32 + nj * 16 + ln;
      float bv = bias[n];
      for (int v = 0; v < 8; ++v) {
        int m = mBase + wr * 64 + mi * 16 + hi * 8 + v;
        float tv = acc[mi][nj][v] + bv;
        if (GELU) tv = 0.5f * tv * (1.0f + erff(tv * 0.70710678118654752f));
        tv *= scale;
        if (RES) tv += res[(size_t)m * N + n];
        if (OUTBF) outb[(size_t)m * N + n] = to_bf16(tv);
        else       outf[(size_t)m * N + n] = tv;
      }
    }
}

// ---------------- position-bias MLP (BN folded), one block per window ----------------
__global__ void posbias_kernel(const float* __restrict__ pos,
                               const float* __restrict__ w1, const float* __restrict__ b1,
                               const float* __restrict__ bg, const float* __restrict__ bb,
                               const float* __restrict__ bm, const float* __restrict__ bvv,
                               const float* __restrict__ w2, const float* __restrict__ b2,
                               float* __restrict__ bias) {
  __shared__ float px[32], py[32];
  __shared__ float a1[16], c1[16], wa[16], wb[16], w2s[128], b2s[8];
  int w = blockIdx.x;
  int t = threadIdx.x;
  if (t < 32) {
    int g = win_row(w, t);
    px[t] = pos[(size_t)g * 2];
    py[t] = pos[(size_t)g * 2 + 1];
  }
  if (t < 16) {
    float a = bg[t] * rsqrtf(bvv[t] + EPSV);
    a1[t] = a;
    c1[t] = bb[t] + a * (b1[t] - bm[t]);   // BN(lin + b1) = a*lin + c1
    wa[t] = w1[t * 2];
    wb[t] = w1[t * 2 + 1];
  }
  if (t < 128) w2s[t] = w2[t];
  if (t < 8)   b2s[t] = b2[t];
  __syncthreads();

  for (int p = t; p < NTOK * NTOK; p += 256) {
    int i = p >> 5, j = p & 31;
    float dx = px[i] - px[j], dy = py[i] - py[j];
    float hv[16];
    for (int o = 0; o < 16; ++o) {
      float h = a1[o] * (wa[o] * dx + wb[o] * dy) + c1[o];
      hv[o] = fmaxf(h, 0.0f);
    }
    for (int hh = 0; hh < NHD; ++hh) {
      float acc = b2s[hh];
      for (int o = 0; o < 16; ++o) acc += w2s[hh * 16 + o] * hv[o];
      bias[(((size_t)w * NHD + hh) * NTOK + i) * NTOK + j] = acc;
    }
  }
}

// ---------------- windowed attention: one wave per (window, head) ----------------
__global__ void attn_kernel(const bf16* __restrict__ Q, const bf16* __restrict__ Km,
                            const bf16* __restrict__ V, const float* __restrict__ Bias,
                            bf16* __restrict__ O) {
  __shared__ __align__(16) bf16 smem[8][2][32 * 40];   // per wave: P, V staging
  int w    = blockIdx.x;
  int wave = threadIdx.x >> 5;
  int lane = threadIdx.x & 31;
  int h    = wave;
  bf16* Pl = smem[wave][0];
  bf16* Vl = smem[wave][1];
  int ln = lane & 15, hi = lane >> 4, kb = hi << 3;

  // Q as A fragments (two 16-row halves), K as B fragments (two 16-key halves);
  // both are K-contiguous over head-dim in memory -> same load pattern.
  v16bf qa[2], kbf[2];
  for (int mi = 0; mi < 2; ++mi) {
    int g = win_row(w, mi * 16 + ln);
    qa[mi] = frag_ld(Q + (size_t)g * CC + h * HD + kb);
  }
  for (int nj = 0; nj < 2; ++nj) {
    int g = win_row(w, nj * 16 + ln);
    kbf[nj] = frag_ld(Km + (size_t)g * CC + h * HD + kb);
  }

  // scores = (Q*scale) . K^T, seeded with relative-position bias as the C matrix
  const float* bp = Bias + (((size_t)w * NHD + h) * NTOK) * NTOK;
  v8f s[2][2];
  for (int mi = 0; mi < 2; ++mi)
    for (int nj = 0; nj < 2; ++nj) {
      v8f c;
      for (int v = 0; v < 8; ++v) {
        int i = mi * 16 + hi * 8 + v;
        int j = nj * 16 + ln;
        c[v] = bp[i * NTOK + j];
      }
      s[mi][nj] = wmma_bf16(qa[mi], kbf[nj], c);
    }

  // stage V rows (keys) into LDS while softmax runs
  {
    int g = win_row(w, lane);
    const bf16* p = V + (size_t)g * CC + h * HD;
    for (int ck = 0; ck < 4; ++ck)
      *(v8bf*)&Vl[lane * 40 + ck * 8] = *(const v8bf*)(p + ck * 8);
  }

  // row softmax: a row's 32 scores live in one 16-lane group at one vgpr slot
  for (int mi = 0; mi < 2; ++mi)
    for (int v = 0; v < 8; ++v) {
      float mx = fmaxf(s[mi][0][v], s[mi][1][v]);
      for (int msk = 1; msk < 16; msk <<= 1) mx = fmaxf(mx, __shfl_xor(mx, msk, 32));
      float e0 = expf(s[mi][0][v] - mx);
      float e1 = expf(s[mi][1][v] - mx);
      float sm = e0 + e1;
      for (int msk = 1; msk < 16; msk <<= 1) sm += __shfl_xor(sm, msk, 32);
      float r = 1.0f / sm;
      int i = mi * 16 + hi * 8 + v;
      Pl[i * 40 + ln]      = to_bf16(e0 * r);
      Pl[i * 40 + 16 + ln] = to_bf16(e1 * r);
    }
  __syncthreads();

  // O = P . V : P as A fragments from LDS, V columns (head-dim) as B fragments
  v16bf pa[2], vb[2];
  for (int mi = 0; mi < 2; ++mi)
    pa[mi] = frag_ld(&Pl[(mi * 16 + ln) * 40 + kb]);
  for (int nj = 0; nj < 2; ++nj)
    for (int i = 0; i < 8; ++i) {
      vb[nj][i]     = Vl[(kb + i) * 40      + nj * 16 + ln];
      vb[nj][i + 8] = Vl[(kb + 16 + i) * 40 + nj * 16 + ln];
    }

  for (int mi = 0; mi < 2; ++mi)
    for (int nj = 0; nj < 2; ++nj) {
      v8f z;
      for (int v = 0; v < 8; ++v) z[v] = 0.0f;
      v8f o = wmma_bf16(pa[mi], vb[nj], z);
      for (int v = 0; v < 8; ++v) {
        int i = mi * 16 + hi * 8 + v;
        int g = win_row(w, i);
        O[(size_t)g * CC + h * HD + nj * 16 + ln] = to_bf16(o[v]);
      }
    }
}

// ---------------- launcher ----------------
extern "C" void kernel_launch(void* const* d_in, const int* in_sizes, int n_in,
                              void* d_out, int out_size, void* d_ws, size_t ws_size,
                              hipStream_t stream) {
  const float* x     = (const float*)d_in[0];
  const float* pos   = (const float*)d_in[1];
  const float* ln1_g = (const float*)d_in[2];
  const float* ln1_b = (const float*)d_in[3];
  const float* Wq    = (const float*)d_in[4];
  const float* bq    = (const float*)d_in[5];
  const float* Wk    = (const float*)d_in[6];
  const float* bk    = (const float*)d_in[7];
  const float* Wv    = (const float*)d_in[8];
  const float* bv    = (const float*)d_in[9];
  const float* Wp    = (const float*)d_in[10];
  const float* bp    = (const float*)d_in[11];
  const float* ln2_g = (const float*)d_in[12];
  const float* ln2_b = (const float*)d_in[13];
  const float* fc1_w = (const float*)d_in[14];
  const float* fc1_b = (const float*)d_in[15];
  const float* fc2_w = (const float*)d_in[16];
  const float* fc2_b = (const float*)d_in[17];
  const float* pe_w1 = (const float*)d_in[18];
  const float* pe_b1 = (const float*)d_in[19];
  const float* bn_g  = (const float*)d_in[20];
  const float* bn_b  = (const float*)d_in[21];
  const float* bn_m  = (const float*)d_in[22];
  const float* bn_v  = (const float*)d_in[23];
  const float* pe_w2 = (const float*)d_in[24];
  const float* pe_b2 = (const float*)d_in[25];

  char* wp = (char*)d_ws;
  auto alloc = [&](size_t bytes) {
    void* r = (void*)wp;
    wp += (bytes + 255) & ~(size_t)255;
    return r;
  };
  bf16* Wq_bf  = (bf16*)alloc((size_t)CC * CC * 2);
  bf16* Wk_bf  = (bf16*)alloc((size_t)CC * CC * 2);
  bf16* Wv_bf  = (bf16*)alloc((size_t)CC * CC * 2);
  bf16* Wp_bf  = (bf16*)alloc((size_t)CC * CC * 2);
  bf16* fc1_bf = (bf16*)alloc((size_t)FF * CC * 2);
  bf16* fc2_bf = (bf16*)alloc((size_t)CC * FF * 2);
  bf16* xn_bf  = (bf16*)alloc((size_t)MTOK * CC * 2);
  bf16* q_bf   = (bf16*)alloc((size_t)MTOK * CC * 2);
  bf16* k_bf   = (bf16*)alloc((size_t)MTOK * CC * 2);
  bf16* v_bf   = (bf16*)alloc((size_t)MTOK * CC * 2);
  bf16* o_bf   = (bf16*)alloc((size_t)MTOK * CC * 2);
  float* biasw = (float*)alloc((size_t)NWIN * NHD * NTOK * NTOK * 4);
  float* x1    = (float*)alloc((size_t)MTOK * CC * 4);
  bf16* xm_bf  = (bf16*)alloc((size_t)MTOK * CC * 2);
  bf16* h1_bf  = (bf16*)alloc((size_t)MTOK * FF * 2);

  // 1) weights -> bf16
  cvt_kernel<<<(CC * CC + 255) / 256, 256, 0, stream>>>(Wq, Wq_bf, CC * CC);
  cvt_kernel<<<(CC * CC + 255) / 256, 256, 0, stream>>>(Wk, Wk_bf, CC * CC);
  cvt_kernel<<<(CC * CC + 255) / 256, 256, 0, stream>>>(Wv, Wv_bf, CC * CC);
  cvt_kernel<<<(CC * CC + 255) / 256, 256, 0, stream>>>(Wp, Wp_bf, CC * CC);
  cvt_kernel<<<(FF * CC + 255) / 256, 256, 0, stream>>>(fc1_w, fc1_bf, FF * CC);
  cvt_kernel<<<(CC * FF + 255) / 256, 256, 0, stream>>>(fc2_w, fc2_bf, CC * FF);

  // 2) LN1
  ln_kernel<<<MTOK / 8, 256, 0, stream>>>(x, ln1_g, ln1_b, xn_bf);

  // 3) QKV projections (Q pre-scaled by 1/sqrt(HD))
  dim3 gC(CC / 128, MTOK / 128);
  gemm_bf16_kernel<false, false, true><<<gC, 256, 0, stream>>>(
      xn_bf, Wq_bf, bq, nullptr, nullptr, q_bf, MTOK, CC, CC, SCL);
  gemm_bf16_kernel<false, false, true><<<gC, 256, 0, stream>>>(
      xn_bf, Wk_bf, bk, nullptr, nullptr, k_bf, MTOK, CC, CC, 1.0f);
  gemm_bf16_kernel<false, false, true><<<gC, 256, 0, stream>>>(
      xn_bf, Wv_bf, bv, nullptr, nullptr, v_bf, MTOK, CC, CC, 1.0f);

  // 4) relative-position bias
  posbias_kernel<<<NWIN, 256, 0, stream>>>(pos, pe_w1, pe_b1, bn_g, bn_b, bn_m,
                                           bn_v, pe_w2, pe_b2, biasw);

  // 5) windowed attention
  attn_kernel<<<NWIN, 256, 0, stream>>>(q_bf, k_bf, v_bf, biasw, o_bf);

  // 6) output projection + residual -> x1 (fp32)
  gemm_bf16_kernel<false, true, false><<<gC, 256, 0, stream>>>(
      o_bf, Wp_bf, bp, x, x1, nullptr, MTOK, CC, CC, 1.0f);

  // 7) LN2
  ln_kernel<<<MTOK / 8, 256, 0, stream>>>(x1, ln2_g, ln2_b, xm_bf);

  // 8) MLP: fc1 + exact GELU -> bf16
  dim3 gF(FF / 128, MTOK / 128);
  gemm_bf16_kernel<true, false, true><<<gF, 256, 0, stream>>>(
      xm_bf, fc1_bf, fc1_b, nullptr, nullptr, h1_bf, MTOK, FF, CC, 1.0f);

  // 9) fc2 + residual -> final output (fp32)
  gemm_bf16_kernel<false, true, false><<<gC, 256, 0, stream>>>(
      h1_bf, fc2_bf, fc2_b, x1, (float*)d_out, nullptr, MTOK, CC, FF, 1.0f);
}
